// LocalTokenAttention_32684701122889
// MI455X (gfx1250) — compile-verified, matrix operand order checked by reading, not run on hardware
//
#include <hip/hip_runtime.h>
#include <hip/hip_bf16.h>

typedef __attribute__((ext_vector_type(16))) _Float16 v16h;
typedef __attribute__((ext_vector_type(8)))  float    v8f;

#define C_DIM 192
#define HWDIM 128
#define NWIN  9
#define STEPW 14
#define S_TOK 256
#define HEADS 6
#define LXS   200   // lx / kb / q row stride in halves
#define VBS   256   // vb (transposed V) row stride in halves
#define WQ_OFF 0
#define WK_OFF (C_DIM*C_DIM)
#define WV_OFF (2*C_DIM*C_DIM)

// Load a 16x32 f16 WMMA fragment (A or B) where the K dimension is contiguous.
// `base` already points at the fixed row/column for this lane; CDNA5 layout:
// lanes 0-15 hold K = [0..7] + [16..23], lanes 16-31 hold K = [8..15] + [24..31].
template <typename P>
__device__ __forceinline__ v16h ldfrag(const P* base, int k0, int hi) {
  v16h f;
#pragma unroll
  for (int j = 0; j < 8; ++j) f[j] = (_Float16)base[k0 + 8 * hi + j];
#pragma unroll
  for (int j = 0; j < 8; ++j) f[j + 8] = (_Float16)base[k0 + 16 + 8 * hi + j];
  return f;
}

// CDNA5 async global->LDS copy (ASYNCcnt-tracked, bypasses VGPRs).
__device__ __forceinline__ void async_copy_b64(unsigned lds_addr,
                                               const float* gaddr) {
  asm volatile("global_load_async_to_lds_b64 %0, %1, off"
               :: "v"(lds_addr), "v"(gaddr) : "memory");
}
__device__ __forceinline__ void wait_async_zero() {
  asm volatile("s_wait_asynccnt 0x0" ::: "memory");
}

__global__ void lta_prep_weights(const float* __restrict__ wq,
                                 const float* __restrict__ wk,
                                 const float* __restrict__ wv,
                                 _Float16* __restrict__ wh) {
  int i = blockIdx.x * blockDim.x + threadIdx.x;
  if (i < C_DIM * C_DIM) {
    wh[WQ_OFF + i] = (_Float16)wq[i];
    wh[WK_OFF + i] = (_Float16)wk[i];
    wh[WV_OFF + i] = (_Float16)wv[i];
  }
}

__global__ void lta_zero(float* __restrict__ o, int n) {
  int i = blockIdx.x * blockDim.x + threadIdx.x;
  if (i < n) o[i] = 0.f;
}

// final: divide by overlap count; starts are 14*i, i=0..8, window 16
__device__ __forceinline__ int cover1d(int p) {
  int imin = (p >= 15) ? ((p - 15 + 13) / 14) : 0;
  int imax = p / 14; if (imax > 8) imax = 8;
  return imax - imin + 1;
}

__global__ void lta_divcount(float* __restrict__ o, int n) {
  int i = blockIdx.x * blockDim.x + threadIdx.x;
  if (i >= n) return;
  int px = i & (HWDIM - 1);
  int py = (i >> 7) & (HWDIM - 1);
  o[i] *= 1.f / (float)(cover1d(py) * cover1d(px));
}

__global__ __launch_bounds__(256, 1)
void lta_win_attn(const float* __restrict__ x,
                  const float* __restrict__ gamma,
                  const float* __restrict__ beta,
                  const _Float16* __restrict__ wh,
                  float* __restrict__ out) {
  extern __shared__ _Float16 smem[];
  _Float16* lx  = smem;                 // [256][LXS] normalized input, later Q
  _Float16* kb  = lx + S_TOK * LXS;     // [256][LXS] K          (phase >= 1)
  _Float16* vb  = kb + S_TOK * LXS;     // [192][VBS] V^T        (phase >= 1)
  _Float16* psm = vb + C_DIM * VBS;     // 8 waves x 16x32 P     (phase >= 1)
  // Phase 0 only: f32 window staging buffer overlapping kb/vb/psm region.
  float* xw = (float*)kb;               // [192][256] channel-major window

  const int win  = blockIdx.x;
  const int b    = win / (NWIN * NWIN);
  const int wi   = win % (NWIN * NWIN);
  const int top  = STEPW * (wi / NWIN);
  const int left = STEPW * (wi % NWIN);

  const int tid  = threadIdx.x;
  const int lane = tid & 31;
  const int wid  = tid >> 5;
  const int l15  = lane & 15;
  const int hi   = lane >> 4;

  const _Float16* wq = wh + WQ_OFF;
  const _Float16* wk = wh + WK_OFF;
  const _Float16* wv = wh + WV_OFF;

  // ---- Phase 0a: async DMA of the whole window into LDS (single read of x).
  // 192ch x 16rows x 8 pixel-pairs = 24576 b64 transfers, 96 per thread.
  {
    const float* xbase = x + (size_t)b * C_DIM * (HWDIM * HWDIM);
    for (int i = 0; i < 96; ++i) {
      const int e   = tid + 256 * i;      // 0..24575
      const int pp  = e & 7;              // pixel pair in row
      const int row = (e >> 3) & 15;
      const int c   = e >> 7;             // 0..191
      const float* g = xbase + ((size_t)c * HWDIM + top + row) * HWDIM
                             + left + pp * 2;
      const unsigned lds =
          (unsigned)(uintptr_t)(xw + c * 256 + row * 16 + pp * 2);
      async_copy_b64(lds, g);
    }
    wait_async_zero();
  }
  __syncthreads();

  // ---- Phase 0b: LayerNorm out of LDS, one token per thread ----
  {
    const int s = tid;
    float sum = 0.f, sq = 0.f;
    for (int c = 0; c < C_DIM; ++c) {
      float v = xw[c * 256 + s];
      sum += v; sq += v * v;
    }
    float mu  = sum * (1.f / C_DIM);
    float var = sq * (1.f / C_DIM) - mu * mu;
    float rs  = rsqrtf(var + 1e-5f);
    for (int c = 0; c < C_DIM; ++c) {
      float v = (xw[c * 256 + s] - mu) * rs * gamma[c] + beta[c];
      lx[s * LXS + c] = (_Float16)v;
    }
  }
  __syncthreads();

  // ---------------- Phase 1: K and V GEMMs (tiles split across waves) ------
  // 16 M-tiles x 12 N-tiles = 192 tiles; each = 6 WMMA K-steps.
  // (kb/vb/psm overwrite the xw staging region -- xw is dead now.)
  for (int tile = wid; tile < 16 * 12; tile += 8) {
    const int mt = tile / 12, nt = tile % 12;
    const _Float16* abase = lx + (mt * 16 + l15) * LXS;
    v8f ak = {}, av = {};
#pragma unroll
    for (int kk = 0; kk < 6; ++kk) {
      v16h a  = ldfrag(abase, kk * 32, hi);
      v16h bk = ldfrag(wk + (nt * 16 + l15) * C_DIM, kk * 32, hi);
      v16h bv = ldfrag(wv + (nt * 16 + l15) * C_DIM, kk * 32, hi);
      ak = __builtin_amdgcn_wmma_f32_16x16x32_f16(false, a, false, bk,
                                                  (short)0, ak, false, false);
      av = __builtin_amdgcn_wmma_f32_16x16x32_f16(false, a, false, bv,
                                                  (short)0, av, false, false);
    }
#pragma unroll
    for (int r = 0; r < 8; ++r) {
      kb[(mt * 16 + hi * 8 + r) * LXS + nt * 16 + l15] = (_Float16)ak[r];
      // V stored transposed: contiguous along token for PV B-fragments
      vb[(nt * 16 + l15) * VBS + mt * 16 + hi * 8 + r] = (_Float16)av[r];
    }
  }
  __syncthreads();

  // ---------------- Phase 2: Q for this wave's 32 rows, overwrite lx -------
#pragma unroll
  for (int mtile = 0; mtile < 2; ++mtile) {
    const int mrow0 = wid * 32 + mtile * 16;
    v8f qa[12];
#pragma unroll
    for (int nt = 0; nt < 12; ++nt) qa[nt] = (v8f){};
#pragma unroll
    for (int kk = 0; kk < 6; ++kk) {
      v16h a = ldfrag(lx + (mrow0 + l15) * LXS, kk * 32, hi);
#pragma unroll
      for (int nt = 0; nt < 12; ++nt) {
        v16h bq = ldfrag(wq + (nt * 16 + l15) * C_DIM, kk * 32, hi);
        qa[nt] = __builtin_amdgcn_wmma_f32_16x16x32_f16(false, a, false, bq,
                                                        (short)0, qa[nt],
                                                        false, false);
      }
    }
    // all reads of these lx rows done (in-wave ordering) -> overwrite with Q
#pragma unroll
    for (int nt = 0; nt < 12; ++nt)
#pragma unroll
      for (int r = 0; r < 8; ++r)
        lx[(mrow0 + hi * 8 + r) * LXS + nt * 16 + l15] = (_Float16)qa[nt][r];
  }
  __syncthreads();

  // ---------------- Phase 3: flash attention per (mtile, head) -------------
  const float scale = 0.07216878364870323f; // 192^-0.5
  _Float16* pws = psm + wid * 512;          // wave-private 16x32 P scratch

  for (int mtile = 0; mtile < 2; ++mtile) {
    const int mrow0 = wid * 32 + mtile * 16;
    for (int h = 0; h < HEADS; ++h) {
      const v16h qf = ldfrag(lx + (mrow0 + l15) * LXS, h * 32, hi);
      v8f o0 = {}, o1 = {};
      float mrow[8], lrow[8];
#pragma unroll
      for (int r = 0; r < 8; ++r) { mrow[r] = -3.0e30f; lrow[r] = 0.f; }

      for (int tp = 0; tp < 8; ++tp) {          // 32 keys per iteration
        v16h kf0 = ldfrag(kb + (tp * 32 +      l15) * LXS, h * 32, hi);
        v16h kf1 = ldfrag(kb + (tp * 32 + 16 + l15) * LXS, h * 32, hi);
        v8f s0 = {}, s1 = {};
        s0 = __builtin_amdgcn_wmma_f32_16x16x32_f16(false, qf, false, kf0,
                                                    (short)0, s0, false, false);
        s1 = __builtin_amdgcn_wmma_f32_16x16x32_f16(false, qf, false, kf1,
                                                    (short)0, s1, false, false);
#pragma unroll
        for (int r = 0; r < 8; ++r) {
          float a = s0[r] * scale;
          float c = s1[r] * scale;
          float mx = fmaxf(a, c);
#pragma unroll
          for (int off = 1; off < 16; off <<= 1)
            mx = fmaxf(mx, __shfl_xor(mx, off, 32));
          float mnew = fmaxf(mrow[r], mx);
          float p0 = __expf(a - mnew);
          float p1 = __expf(c - mnew);
          float ps = p0 + p1;
#pragma unroll
          for (int off = 1; off < 16; off <<= 1)
            ps += __shfl_xor(ps, off, 32);
          float corr = __expf(mrow[r] - mnew);
          lrow[r] = lrow[r] * corr + ps;
          mrow[r] = mnew;
          o0[r] *= corr;
          o1[r] *= corr;
          pws[(hi * 8 + r) * 32 + l15]      = (_Float16)p0;
          pws[(hi * 8 + r) * 32 + 16 + l15] = (_Float16)p1;
        }
        v16h pf  = ldfrag(pws + l15 * 32, 0, hi);
        v16h vf0 = ldfrag(vb + (h * 32 +      l15) * VBS, tp * 32, hi);
        v16h vf1 = ldfrag(vb + (h * 32 + 16 + l15) * VBS, tp * 32, hi);
        o0 = __builtin_amdgcn_wmma_f32_16x16x32_f16(false, pf, false, vf0,
                                                    (short)0, o0, false, false);
        o1 = __builtin_amdgcn_wmma_f32_16x16x32_f16(false, pf, false, vf1,
                                                    (short)0, o1, false, false);
      }

      // epilogue: normalize and scatter-add into output image
#pragma unroll
      for (int r = 0; r < 8; ++r) {
        const int stok = mrow0 + hi * 8 + r;     // token in window [0,256)
        const float inv = 1.f / lrow[r];
        const int py = top  + (stok >> 4);
        const int px = left + (stok & 15);
        const int c0 = h * 32 + l15;
        float* o_base = out + (((size_t)b * C_DIM + c0) * HWDIM + py) * HWDIM + px;
        atomicAdd(o_base, o0[r] * inv);
        atomicAdd(o_base + (size_t)16 * HWDIM * HWDIM, o1[r] * inv);
      }
    }
  }
}

extern "C" void kernel_launch(void* const* d_in, const int* in_sizes, int n_in,
                              void* d_out, int out_size, void* d_ws, size_t ws_size,
                              hipStream_t stream) {
  const float* x     = (const float*)d_in[0];
  const float* Wq    = (const float*)d_in[1];
  const float* Wk    = (const float*)d_in[2];
  const float* Wv    = (const float*)d_in[3];
  const float* gamma = (const float*)d_in[4];
  const float* beta  = (const float*)d_in[5];
  float* out = (float*)d_out;
  _Float16* wh = (_Float16*)d_ws;

  // 1) f32 -> f16 weight conversion into workspace
  lta_prep_weights<<<(C_DIM * C_DIM + 255) / 256, 256, 0, stream>>>(Wq, Wk, Wv, wh);
  // 2) zero the output accumulator
  lta_zero<<<(out_size + 255) / 256, 256, 0, stream>>>(out, out_size);
  // 3) fused LN + QKV + flash attention + scatter-add (one block per window)
  const size_t shmem =
      (size_t)(2 * S_TOK * LXS + C_DIM * VBS + 8 * 512) * sizeof(_Float16);
  lta_win_attn<<<4 * NWIN * NWIN, 256, shmem, stream>>>(x, gamma, beta, wh, out);
  // 4) divide by overlap counts
  lta_divcount<<<(out_size + 255) / 256, 256, 0, stream>>>(out, out_size);
}
